// EvoformerBlock_3358664425717
// MI455X (gfx1250) — matrix operand order repaired and checked by reading, not compiled, not run on hardware
//
#include <hip/hip_runtime.h>
#include <cstddef>

// ---------------------------------------------------------------------------
// Evoformer block for MI455X (gfx1250, wave32, WMMA bf16 path + async LDS)
#define S_   128
#define R_   256
#define CM_  256
#define CZ_  128
#define HM_  8
#define HZ_  4
#define SR_  (S_ * R_)                 // 32768
#define RR_  (R_ * R_)                 // 65536
#define SLAB ((size_t)SR_ * CM_)       // 8,388,608 elements

typedef __attribute__((ext_vector_type(16))) __bf16 v16bf;
typedef __attribute__((ext_vector_type(8)))  __bf16 v8bf;
typedef __attribute__((ext_vector_type(4)))  __bf16 v4bf;
typedef __attribute__((ext_vector_type(8)))  float  v8f;

static __device__ __forceinline__ v16bf cat8(v8bf a, v8bf b) {
  return __builtin_shufflevector(a, b, 0, 1, 2, 3, 4, 5, 6, 7, 8, 9, 10, 11,
                                 12, 13, 14, 15);
}
static __device__ __forceinline__ v8f vzero8() {
  v8f z = {0.f, 0.f, 0.f, 0.f, 0.f, 0.f, 0.f, 0.f};
  return z;
}
static __device__ __forceinline__ v8bf bzero8() {
  v8bf z;
#pragma unroll
  for (int i = 0; i < 8; ++i) z[i] = (__bf16)0.f;
  return z;
}
static __device__ __forceinline__ float sigm(float x) {
  return 1.f / (1.f + __expf(-x));
}
static __device__ __forceinline__ v8f wmma_bf16(v16bf a, v16bf b, v8f c) {
  return __builtin_amdgcn_wmma_f32_16x16x32_bf16(false, a, false, b, (short)0,
                                                 c, false, false);
}
// Async global->LDS copy (16B per lane), tracked by ASYNCcnt (CDNA5 ISA).
static __device__ __forceinline__ void async_b128(unsigned int lds_off,
                                                  const __bf16* g) {
  const unsigned long long ga = (unsigned long long)(size_t)g;
  asm volatile("global_load_async_to_lds_b128 %0, %1, off" ::"v"(lds_off),
               "v"(ga)
               : "memory");
}
static __device__ __forceinline__ void wait_async() {
  asm volatile("s_wait_asynccnt 0x0" ::: "memory");
}
static __device__ __forceinline__ unsigned int lds_off(const void* p) {
  return (unsigned int)(size_t)p;
}

// ---------------------------------------------------------------------------
// Wide fragment loads (CDNA5 wave32 layouts, CDNA5 ISA §7.12.2).
static __device__ __forceinline__ v16bf frag_a_lds(const __bf16* base,
                                                   int rowStride, int lane) {
  const int hf = (lane >> 4) & 1, m = lane & 15;
  const __bf16* p = base + m * rowStride + 8 * hf;
  v8bf lo = *(const v8bf*)p;
  v8bf hi = *(const v8bf*)(p + 16);
  return cat8(lo, hi);
}
static __device__ __forceinline__ v16bf frag_b_ldsT(const __bf16* base,
                                                    int colStride, int lane) {
  const int kh = (lane >> 4) & 1, n = lane & 15;
  const __bf16* p = base + n * colStride + 8 * kh;
  v8bf lo = *(const v8bf*)p;
  v8bf hi = *(const v8bf*)(p + 16);
  return cat8(lo, hi);
}
static __device__ __forceinline__ v16bf frag_a_gbf(const __bf16* base,
                                                   int rowStride, int lane) {
  const int hf = (lane >> 4) & 1, m = lane & 15;
  const __bf16* p = base + (size_t)m * rowStride + 8 * hf;
  v8bf lo = *(const v8bf*)p;
  v8bf hi = *(const v8bf*)(p + 16);
  return cat8(lo, hi);
}

// ---------------------------------------------------------------------------
// Weight convert + transpose: wt[n*K+k] = (bf16) w[k*N+n]
__global__ void k_wcvt(const float* __restrict__ w, __bf16* __restrict__ wt,
                       int K, int N) {
  const size_t i = (size_t)blockIdx.x * 256 + threadIdx.x;
  if (i >= (size_t)K * N) return;
  const int k = (int)(i % K);
  const size_t n = i / K;
  wt[i] = (__bf16)w[(size_t)k * N + n];
}

// ---------------------------------------------------------------------------
// LayerNorm, one row per wave, f32 in -> bf16 out.
template <int C>
__global__ __launch_bounds__(128) void k_layernorm(
    const float* __restrict__ x, const float* __restrict__ g,
    const float* __restrict__ b, __bf16* __restrict__ y) {
  constexpr int CPL = C / 32;
  const int wave = threadIdx.x >> 5, lane = threadIdx.x & 31;
  const size_t row = (size_t)blockIdx.x * 4 + wave;
  const float* xr = x + row * C + lane * CPL;
  float vals[CPL];
  float s = 0.f, s2 = 0.f;
#pragma unroll
  for (int q = 0; q < CPL; ++q) {
    vals[q] = xr[q];
    s += vals[q];
    s2 += vals[q] * vals[q];
  }
#pragma unroll
  for (int msk = 1; msk < 32; msk <<= 1) {
    s += __shfl_xor(s, msk, 32);
    s2 += __shfl_xor(s2, msk, 32);
  }
  const float mu = s / (float)C;
  const float var = s2 / (float)C - mu * mu;
  const float rstd = rsqrtf(var + 1e-5f);
  __bf16* yr = y + row * C + lane * CPL;
  const float* gr = g + lane * CPL;
  const float* br = b + lane * CPL;
#pragma unroll
  for (int q = 0; q < CPL; ++q)
    yr[q] = (__bf16)((vals[q] - mu) * rstd * gr[q] + br[q]);
}

// ---------------------------------------------------------------------------
// GEMM: C[M,N] (+)= epilogue( A[M,K] * BT[N,K]^T ).  A, BT bf16; acc f32.
// 128x64 block tile; each of 4 waves owns a 32-row strip -> 8 WMMA per step
// per 6 fragment loads.  Fast path: async global->LDS copies, double
// buffered, one barrier per K-step.
// act: 0 none, 1 relu, 2 sigmoid.  mul: f32 gate.  transc: write C^T.
// outbf: write bf16 (else f32; addto only valid with f32).
__global__ __launch_bounds__(128) void k_gemm(
    const __bf16* __restrict__ A, const __bf16* __restrict__ BT,
    void* __restrict__ Cv, const float* __restrict__ bias,
    const float* __restrict__ mul, int M, int N, int K, int act, int addto,
    int transc, int outbf) {
  __shared__ __align__(16) __bf16 As[2][128 * 32];  // [m][k] 8KB each
  __shared__ __align__(16) __bf16 Bs[2][64 * 32];   // [n][k] 4KB each
  const int tid = threadIdx.x, wave = tid >> 5, lane = tid & 31;
  const int m0 = blockIdx.y * 128, n0 = blockIdx.x * 64;
  const int rr = tid >> 1, ch0 = (tid & 1) * 16;  // B: row + 16-elem half

  v8f acc[2][4];
#pragma unroll
  for (int mt = 0; mt < 2; ++mt)
#pragma unroll
    for (int i = 0; i < 4; ++i) acc[mt][i] = vzero8();

  auto compute = [&](int buf) {
    v16bf af0 = frag_a_lds(&As[buf][(wave * 32) * 32], 32, lane);
    v16bf af1 = frag_a_lds(&As[buf][(wave * 32 + 16) * 32], 32, lane);
#pragma unroll
    for (int cn = 0; cn < 4; ++cn) {
      v16bf bf = frag_b_ldsT(&Bs[buf][(cn * 16) * 32], 32, lane);
      acc[0][cn] = wmma_bf16(af0, bf, acc[0][cn]);
      acc[1][cn] = wmma_bf16(af1, bf, acc[1][cn]);
    }
  };

  const bool fast = (m0 + 128 <= M) && (n0 + 64 <= N);
  if (fast) {
    const __bf16* ga = A + (size_t)(m0 + tid) * K;  // one 64B row per thread
    const __bf16* gb = BT + (size_t)(n0 + rr) * K + ch0;
    const unsigned int la = lds_off(&As[0][tid * 32]);
    const unsigned int lb = lds_off(&Bs[0][rr * 32 + ch0]);
#pragma unroll
    for (int q = 0; q < 4; ++q) async_b128(la + 16 * q, ga + 8 * q);
    async_b128(lb, gb);
    async_b128(lb + 16, gb + 8);
    wait_async();
    __syncthreads();
    int buf = 0;
    for (int kt = 0; kt < K; kt += 32) {
      const bool more = (kt + 32) < K;
      if (more) {  // issue next tile into the other buffer
        const unsigned int nb = (buf ^ 1);
        const unsigned int boa = nb ? 8192u : 0u;
        const unsigned int bob = nb ? 4096u : 0u;
#pragma unroll
        for (int q = 0; q < 4; ++q)
          async_b128(la + boa + 16 * q, ga + kt + 32 + 8 * q);
        async_b128(lb + bob, gb + kt + 32);
        async_b128(lb + bob + 16, gb + kt + 40);
      }
      compute(buf);
      if (more) wait_async();
      __syncthreads();
      buf ^= 1;
    }
  } else {
    const bool rowA = (m0 + tid < M), rowB = (n0 + rr < N);
    for (int kt = 0; kt < K; kt += 32) {
      v8bf a[4], b0 = bzero8(), b1 = bzero8();
#pragma unroll
      for (int q = 0; q < 4; ++q) a[q] = bzero8();
      if (rowA) {
        const __bf16* s = A + (size_t)(m0 + tid) * K + kt;
#pragma unroll
        for (int q = 0; q < 4; ++q) a[q] = *(const v8bf*)(s + 8 * q);
      }
      if (rowB) {
        const __bf16* s = BT + (size_t)(n0 + rr) * K + kt + ch0;
        b0 = *(const v8bf*)s;
        b1 = *(const v8bf*)(s + 8);
      }
#pragma unroll
      for (int q = 0; q < 4; ++q) *(v8bf*)&As[0][tid * 32 + 8 * q] = a[q];
      *(v8bf*)&Bs[0][rr * 32 + ch0] = b0;
      *(v8bf*)&Bs[0][rr * 32 + ch0 + 8] = b1;
      __syncthreads();
      compute(0);
      __syncthreads();
    }
  }

  const int mb = 8 * (lane >> 4), nl = lane & 15;
  float* Cf = (float*)Cv;
  __bf16* Cb = (__bf16*)Cv;
#pragma unroll
  for (int mt = 0; mt < 2; ++mt)
#pragma unroll
    for (int cn = 0; cn < 4; ++cn)
#pragma unroll
      for (int r = 0; r < 8; ++r) {
        const int m = m0 + wave * 32 + mt * 16 + r + mb;
        const int n = n0 + cn * 16 + nl;
        if (m < M && n < N) {
          float v = acc[mt][cn][r];
          if (bias) v += bias[n];
          if (act == 1) v = fmaxf(v, 0.f);
          else if (act == 2) v = sigm(v);
          if (mul) v *= mul[(size_t)m * N + n];
          const size_t idx =
              transc ? ((size_t)n * M + m) : ((size_t)m * N + n);
          if (outbf) Cb[idx] = (__bf16)v;
          else if (addto) Cf[idx] += v;
          else Cf[idx] = v;
        }
      }
}

// ---------------------------------------------------------------------------
// Fused softmax attention (D = 32).  Q/K/V bf16 [B,N,H,32]; O f32.
// bias (optional): f32 [H,N,N], broadcast over B.
template <int N>
__global__ __launch_bounds__(128) void k_attention(
    const __bf16* __restrict__ Q, const __bf16* __restrict__ Kx,
    const __bf16* __restrict__ V, const float* __restrict__ bias,
    float* __restrict__ O, int B, int H) {
  constexpr int NT = N / 16;
  constexpr int KC = N / 32;
  __shared__ __align__(16) __bf16 Ks[256 * 32];   // [n][d]
  __shared__ __align__(16) __bf16 VsT[32 * 256];  // [d][n]
  __shared__ __align__(16) __bf16 Pb[4][16 * 256];

  const int bh = blockIdx.x, b = bh / H, h = bh % H;
  const int tid = threadIdx.x, wave = tid >> 5, lane = tid & 31;

  for (int u = tid; u < N * 2; u += 128) {
    const int n = u >> 1, dh = (u & 1) * 16;
    const size_t gi = (((size_t)b * N + n) * H + h) * 32 + dh;
    async_b128(lds_off(&Ks[n * 32 + dh]), Kx + gi);  // K: straight copy
    async_b128(lds_off(&Ks[n * 32 + dh + 8]), Kx + gi + 8);
    v8bf v0 = *(const v8bf*)(V + gi);                // V: transpose
    v8bf v1 = *(const v8bf*)(V + gi + 8);
#pragma unroll
    for (int q = 0; q < 8; ++q) {
      VsT[(dh + q) * 256 + n] = v0[q];
      VsT[(dh + 8 + q) * 256 + n] = v1[q];
    }
  }
  wait_async();
  __syncthreads();

  const int qbase = blockIdx.y * 64 + wave * 16;
  const int mb = 8 * (lane >> 4), nl = lane & 15;
  const float* bh_bias = bias ? bias + (size_t)h * N * N : nullptr;

  v16bf qf = frag_a_gbf(&Q[(((size_t)b * N + qbase) * H + h) * 32], H * 32,
                        lane);

  v8f acc[NT];
#pragma unroll
  for (int t = 0; t < NT; ++t) {
    v16bf bf = frag_b_ldsT(&Ks[(t * 16) * 32], 32, lane);
    acc[t] = wmma_bf16(qf, bf, vzero8());
  }

  const float scale = 0.17677669529663687f;  // 1/sqrt(32)
#pragma unroll
  for (int r = 0; r < 8; ++r) {
    const int m = r + mb, q = qbase + m;
    float mx = -3.0e38f;
#pragma unroll
    for (int t = 0; t < NT; ++t) {
      float v = acc[t][r] * scale;
      if (bh_bias) v += bh_bias[(size_t)q * N + (t * 16 + nl)];
      acc[t][r] = v;
      mx = fmaxf(mx, v);
    }
    mx = fmaxf(mx, __shfl_xor(mx, 1, 32));
    mx = fmaxf(mx, __shfl_xor(mx, 2, 32));
    mx = fmaxf(mx, __shfl_xor(mx, 4, 32));
    mx = fmaxf(mx, __shfl_xor(mx, 8, 32));
    float sum = 0.f;
#pragma unroll
    for (int t = 0; t < NT; ++t) {
      const float e = __expf(acc[t][r] - mx);
      acc[t][r] = e;
      sum += e;
    }
    sum += __shfl_xor(sum, 1, 32);
    sum += __shfl_xor(sum, 2, 32);
    sum += __shfl_xor(sum, 4, 32);
    sum += __shfl_xor(sum, 8, 32);
    const float inv = 1.f / sum;
#pragma unroll
    for (int t = 0; t < NT; ++t)
      Pb[wave][m * 256 + t * 16 + nl] = (__bf16)(acc[t][r] * inv);
  }
  __syncthreads();

  v8f oa[2] = {vzero8(), vzero8()};
#pragma unroll
  for (int kc = 0; kc < KC; ++kc) {
    v16bf pf = frag_a_lds(&Pb[wave][kc * 32], 256, lane);
#pragma unroll
    for (int dt = 0; dt < 2; ++dt) {
      v16bf vf = frag_b_ldsT(&VsT[(dt * 16) * 256 + kc * 32], 256, lane);
      oa[dt] = wmma_bf16(pf, vf, oa[dt]);
    }
  }
#pragma unroll
  for (int dt = 0; dt < 2; ++dt)
#pragma unroll
    for (int r = 0; r < 8; ++r) {
      const int q = qbase + r + mb;
      O[(((size_t)b * N + q) * H + h) * 32 + dt * 16 + nl] = oa[dt][r];
    }
}

// ---------------------------------------------------------------------------
// Triangle multiply: x[i,j,c] = sum_k a[i,k,c]*b[j,k,c] (outgoing) or
// sum_k a[k,i,c]*b[k,j,c] (incoming).  a,b bf16 [R,R,CZ]; x f32.
__global__ __launch_bounds__(128) void k_tri(const __bf16* __restrict__ a,
                                             const __bf16* __restrict__ b,
                                             float* __restrict__ x,
                                             int outgoing) {
  __shared__ __align__(16) __bf16 Aas[4][16 * 32];  // [c][m][k]
  __shared__ __align__(16) __bf16 Bbs[4][16 * 32];  // [c][n][k]
  const int tid = threadIdx.x, wave = tid >> 5, lane = tid & 31;
  const int c0 = blockIdx.x * 4;
  const int i0 = blockIdx.y * 16, j0 = blockIdx.z * 16;
  v8f acc = vzero8();
  for (int ks = 0; ks < 8; ++ks) {
    const int k0 = ks * 32;
    for (int u = tid; u < 512; u += 128) {
      const int m = u >> 5, k = u & 31;
      const size_t ga = outgoing ? ((size_t)(i0 + m) * R_ + k0 + k) * CZ_ + c0
                                 : ((size_t)(k0 + k) * R_ + i0 + m) * CZ_ + c0;
      const size_t gb = outgoing ? ((size_t)(j0 + m) * R_ + k0 + k) * CZ_ + c0
                                 : ((size_t)(k0 + k) * R_ + j0 + m) * CZ_ + c0;
      v4bf va = *(const v4bf*)(a + ga);
      v4bf vb = *(const v4bf*)(b + gb);
#pragma unroll
      for (int cl = 0; cl < 4; ++cl) {
        Aas[cl][m * 32 + k] = va[cl];
        Bbs[cl][m * 32 + k] = vb[cl];
      }
    }
    __syncthreads();
    v16bf af = frag_a_lds(Aas[wave], 32, lane);
    v16bf bf = frag_b_ldsT(Bbs[wave], 32, lane);
    acc = wmma_bf16(af, bf, acc);
    __syncthreads();
  }
  const int mb = 8 * (lane >> 4), nl = lane & 15;
  const int c = c0 + wave;
#pragma unroll
  for (int r = 0; r < 8; ++r)
    x[((size_t)(i0 + r + mb) * R_ + j0 + nl) * CZ_ + c] = acc[r];
}

// ---------------------------------------------------------------------------
// OPM stage 1: P[(iloc*R + j), c*32+d] = (1/S) * sum_s a[s,i0+iloc,c]*b[s,j,d]
// a,b bf16 [S,R,32]; P bf16.
__global__ __launch_bounds__(128) void k_opm1(const __bf16* __restrict__ a,
                                              const __bf16* __restrict__ b,
                                              __bf16* __restrict__ P, int i0) {
  __shared__ __align__(16) __bf16 Acs[32 * 32];
  __shared__ __align__(16) __bf16 Bop[4][32 * 32];
  const int tid = threadIdx.x, wave = tid >> 5, lane = tid & 31;
  const int iloc = blockIdx.x;
  const int j0 = blockIdx.y * 4;
  const int i = i0 + iloc;
  v8f acc[2][2] = {{vzero8(), vzero8()}, {vzero8(), vzero8()}};
  for (int ks = 0; ks < 4; ++ks) {
    const int k0 = ks * 32;
    for (int u = tid; u < 320; u += 128) {
      if (u < 64) {
        const int k = u >> 1, ch = (u & 1) * 16;
        const __bf16* s = &a[((size_t)(k0 + k) * R_ + i) * 32 + ch];
        v8bf s0 = *(const v8bf*)s, s1 = *(const v8bf*)(s + 8);
#pragma unroll
        for (int q = 0; q < 8; ++q) {
          Acs[(ch + q) * 32 + k] = s0[q];
          Acs[(ch + 8 + q) * 32 + k] = s1[q];
        }
      } else {
        const int v = u - 64;
        const int jw = v >> 6, k = (v >> 1) & 31, dh = (v & 1) * 16;
        const __bf16* s = &b[((size_t)(k0 + k) * R_ + j0 + jw) * 32 + dh];
        v8bf s0 = *(const v8bf*)s, s1 = *(const v8bf*)(s + 8);
#pragma unroll
        for (int q = 0; q < 8; ++q) {
          Bop[jw][(dh + q) * 32 + k] = s0[q];
          Bop[jw][(dh + 8 + q) * 32 + k] = s1[q];
        }
      }
    }
    __syncthreads();
#pragma unroll
    for (int mt = 0; mt < 2; ++mt) {
      v16bf af = frag_a_lds(&Acs[(mt * 16) * 32], 32, lane);
#pragma unroll
      for (int nt = 0; nt < 2; ++nt) {
        v16bf bf = frag_b_ldsT(&Bop[wave][(nt * 16) * 32], 32, lane);
        acc[mt][nt] = wmma_bf16(af, bf, acc[mt][nt]);
      }
    }
    __syncthreads();
  }
  const int mb = 8 * (lane >> 4), nl = lane & 15;
  const float inv = 1.0f / (float)S_;
  const int j = j0 + wave;
#pragma unroll
  for (int mt = 0; mt < 2; ++mt)
#pragma unroll
    for (int nt = 0; nt < 2; ++nt)
#pragma unroll
      for (int r = 0; r < 8; ++r) {
        const int c = mt * 16 + r + mb, d = nt * 16 + nl;
        P[((size_t)iloc * R_ + j) * 1024 + c * 32 + d] =
            (__bf16)(acc[mt][nt][r] * inv);
      }
}

// ---------------------------------------------------------------------------
__global__ void k_permute01(float* __restrict__ dst,
                            const float* __restrict__ src, int D0, int D1,
                            int C) {
  const size_t i = (size_t)blockIdx.x * 256 + threadIdx.x;
  if (i >= (size_t)D0 * D1 * C) return;
  const int c = (int)(i % C);
  const size_t t = i / C;
  const int d1 = (int)(t % D1);
  const int d0 = (int)(t / D1);
  dst[((size_t)d1 * D0 + d0) * C + c] = src[i];
}
__global__ void k_add_perm(float* __restrict__ dst,
                           const float* __restrict__ src, int D0, int D1,
                           int C) {
  const size_t i = (size_t)blockIdx.x * 256 + threadIdx.x;
  if (i >= (size_t)D0 * D1 * C) return;
  const int c = (int)(i % C);
  const size_t t = i / C;
  const int d1 = (int)(t % D1);
  const int d0 = (int)(t / D1);
  dst[i] += src[((size_t)d1 * D0 + d0) * C + c];
}

// ---------------------------------------------------------------------------
struct TriP {
  const float *lng, *lnb, *ln2g, *ln2b, *paw, *pab, *gaw, *gab, *pbw, *pbb,
      *gbw, *gbb, *ow, *ob, *gw, *gb;
};
struct TAP {
  const float *qw, *kw, *vw, *gw, *gb, *ow, *ob, *lng, *lnb, *bw;
};

extern "C" void kernel_launch(void* const* d_in, const int* in_sizes, int n_in,
                              void* d_out, int out_size, void* d_ws,
                              size_t ws_size, hipStream_t stream) {
  (void)in_sizes; (void)n_in; (void)out_size; (void)ws_size;
  const float* in_m = (const float*)d_in[0];
  const float* in_z = (const float*)d_in[1];
  int pi = 2;
  auto nextp = [&]() { return (const float*)d_in[pi++]; };

  const float *r_qw = nextp(), *r_kw = nextp(), *r_vw = nextp(),
              *r_gw = nextp(), *r_gb = nextp(), *r_ow = nextp(),
              *r_ob = nextp(), *r_lnmg = nextp(), *r_lnmb = nextp(),
              *r_lnzg = nextp(), *r_lnzb = nextp(), *r_bw = nextp();
  const float *c_qw = nextp(), *c_kw = nextp(), *c_vw = nextp(),
              *c_gw = nextp(), *c_gb = nextp(), *c_ow = nextp(),
              *c_ob = nextp(), *c_lng = nextp(), *c_lnb = nextp();
  const float *t_lng = nextp(), *t_lnb = nextp(), *t_w1 = nextp(),
              *t_b1 = nextp(), *t_w2 = nextp(), *t_b2 = nextp();
  const float *op_lng = nextp(), *op_lnb = nextp(), *op_aw = nextp(),
              *op_ab = nextp(), *op_bw = nextp(), *op_bb = nextp(),
              *op_ow = nextp(), *op_ob = nextp();
  auto readTri = [&]() {
    TriP t;
    t.lng = nextp(); t.lnb = nextp(); t.ln2g = nextp(); t.ln2b = nextp();
    t.paw = nextp(); t.pab = nextp(); t.gaw = nextp(); t.gab = nextp();
    t.pbw = nextp(); t.pbb = nextp(); t.gbw = nextp(); t.gbb = nextp();
    t.ow = nextp(); t.ob = nextp(); t.gw = nextp(); t.gb = nextp();
    return t;
  };
  TriP tro = readTri(), tri = readTri();
  auto readTA = [&]() {
    TAP t;
    t.qw = nextp(); t.kw = nextp(); t.vw = nextp(); t.gw = nextp();
    t.gb = nextp(); t.ow = nextp(); t.ob = nextp(); t.lng = nextp();
    t.lnb = nextp(); t.bw = nextp();
    return t;
  };
  TAP tas = readTA(), tae = readTA();
  const float *pt_lng = nextp(), *pt_lnb = nextp(), *pt_w1 = nextp(),
              *pt_b1 = nextp(), *pt_w2 = nextp(), *pt_b2 = nextp();

  float* W = (float*)d_ws;
  size_t off = 0;
  auto wsalloc = [&](size_t nf) { float* p = W + off; off += nf; return p; };
  __bf16* MN = (__bf16*)wsalloc(SLAB / 2 + 64);
  __bf16* ZN = (__bf16*)wsalloc(SLAB / 2 + 64);
  __bf16* Qb = (__bf16*)wsalloc(SLAB / 2 + 64);
  __bf16* Kb = (__bf16*)wsalloc(SLAB / 2 + 64);
  __bf16* Vb = (__bf16*)wsalloc(SLAB / 2 + 64);
  __bf16* Gb = (__bf16*)wsalloc(SLAB / 2 + 64);
  __bf16* Hb = (__bf16*)wsalloc(SLAB * 2 + 64);           // hidden
  __bf16* Ab = (__bf16*)wsalloc((size_t)SR_ * 16 + 64);   // opm a
  __bf16* Bb = (__bf16*)wsalloc((size_t)SR_ * 16 + 64);   // opm b
  __bf16* Pb = (__bf16*)wsalloc((size_t)8 * R_ * 1024 + 64);
  float* Ob = wsalloc(SLAB);
  float* T1 = wsalloc(SLAB);
  float* T2 = wsalloc(SLAB);
  float* MT = wsalloc(SLAB);
  float* BIAS = wsalloc((size_t)RR_ * HM_);
  __bf16* WPOOL = (__bf16*)wsalloc(2 * 1024 * 1024);      // 4M bf16 weights
  size_t woff = 0;
  auto cvtw = [&](const float* w, int K, int N) {
    __bf16* dst = WPOOL + woff;
    woff += (size_t)K * N;
    const size_t tot = (size_t)K * N;
    k_wcvt<<<(unsigned)((tot + 255) / 256), 256, 0, stream>>>(w, dst, K, N);
    return dst;
  };

  float* out_m = (float*)d_out;
  float* out_z = out_m + SLAB;
  hipMemcpyAsync(out_m, in_m, SLAB * sizeof(float), hipMemcpyDeviceToDevice,
                 stream);
  hipMemcpyAsync(out_z, in_z, SLAB * sizeof(float), hipMemcpyDeviceToDevice,
                 stream);

  // ---- convert all weights to bf16, pre-transposed [N][K] ----
  __bf16 *wr_q = cvtw(r_qw, CM_, 256), *wr_k = cvtw(r_kw, CM_, 256),
         *wr_v = cvtw(r_vw, CM_, 256), *wr_g = cvtw(r_gw, CM_, 256),
         *wr_o = cvtw(r_ow, 256, CM_), *wr_b = cvtw(r_bw, CZ_, HM_);
  __bf16 *wc_q = cvtw(c_qw, CM_, 256), *wc_k = cvtw(c_kw, CM_, 256),
         *wc_v = cvtw(c_vw, CM_, 256), *wc_g = cvtw(c_gw, CM_, 256),
         *wc_o = cvtw(c_ow, 256, CM_);
  __bf16 *wt_1 = cvtw(t_w1, CM_, 4 * CM_), *wt_2 = cvtw(t_w2, 4 * CM_, CM_);
  __bf16 *wo_a = cvtw(op_aw, CM_, 32), *wo_b = cvtw(op_bw, CM_, 32),
         *wo_o = cvtw(op_ow, 1024, CZ_);
  auto cvtTri = [&](const TriP& p) {
    struct { __bf16 *pa, *ga, *pb, *gb, *o, *g; } w;
    w.pa = cvtw(p.paw, CZ_, CZ_); w.ga = cvtw(p.gaw, CZ_, CZ_);
    w.pb = cvtw(p.pbw, CZ_, CZ_); w.gb = cvtw(p.gbw, CZ_, CZ_);
    w.o = cvtw(p.ow, CZ_, CZ_);   w.g = cvtw(p.gw, CZ_, CZ_);
    return w;
  };
  auto wtro = cvtTri(tro), wtri = cvtTri(tri);
  auto cvtTA = [&](const TAP& p) {
    struct { __bf16 *q, *k, *v, *g, *o, *b; } w;
    w.q = cvtw(p.qw, CZ_, 128); w.k = cvtw(p.kw, CZ_, 128);
    w.v = cvtw(p.vw, CZ_, 128); w.g = cvtw(p.gw, CZ_, 128);
    w.o = cvtw(p.ow, 128, CZ_); w.b = cvtw(p.bw, CZ_, HZ_);
    return w;
  };
  auto wtas = cvtTA(tas), wtae = cvtTA(tae);
  __bf16 *wp_1 = cvtw(pt_w1, CZ_, 4 * CZ_), *wp_2 = cvtw(pt_w2, 4 * CZ_, CZ_);

  auto LN = [&](const float* x, const float* g, const float* b, __bf16* y,
                int rows, int C) {
    if (C == 256)
      k_layernorm<256><<<rows / 4, 128, 0, stream>>>(x, g, b, y);
    else
      k_layernorm<128><<<rows / 4, 128, 0, stream>>>(x, g, b, y);
  };
  auto GEMM = [&](const __bf16* A, const __bf16* Bt, void* Cm,
                  const float* bias, const float* mul, int M, int N, int K,
                  int act, int addto, int outbf) {
    dim3 g((N + 63) / 64, (M + 127) / 128);
    k_gemm<<<g, 128, 0, stream>>>(A, Bt, Cm, bias, mul, M, N, K, act, addto,
                                  0, outbf);
  };
  auto GEMMT = [&](const __bf16* A, const __bf16* Bt, float* Cm, int M, int N,
                   int K) {
    dim3 g((N + 63) / 64, (M + 127) / 128);
    k_gemm<<<g, 128, 0, stream>>>(A, Bt, Cm, nullptr, nullptr, M, N, K, 0, 0,
                                  1, 0);
  };
  const int PBLK = (int)(SLAB / 256);

  // ================= MSA row attention (pair bias) =================
  LN(out_m, r_lnmg, r_lnmb, MN, SR_, CM_);
  LN(out_z, r_lnzg, r_lnzb, ZN, RR_, CZ_);
  GEMMT(ZN, wr_b, BIAS, RR_, HM_, CZ_);  // BIAS[h][q][k]
  GEMM(MN, wr_q, Qb, nullptr, nullptr, SR_, 256, CM_, 0, 0, 1);
  GEMM(MN, wr_k, Kb, nullptr, nullptr, SR_, 256, CM_, 0, 0, 1);
  GEMM(MN, wr_v, Vb, nullptr, nullptr, SR_, 256, CM_, 0, 0, 1);
  k_attention<256><<<dim3(S_ * HM_, R_ / 64), 128, 0, stream>>>(
      Qb, Kb, Vb, BIAS, Ob, S_, HM_);
  GEMM(MN, wr_g, Gb, r_gb, Ob, SR_, 256, CM_, 2, 0, 1);  // og = sig(g)*o
  GEMM(Gb, wr_o, out_m, r_ob, nullptr, SR_, CM_, 256, 0, 1, 0);

  // ================= MSA column attention =================
  k_permute01<<<PBLK, 256, 0, stream>>>(MT, out_m, S_, R_, CM_);
  LN(MT, c_lng, c_lnb, MN, SR_, CM_);
  GEMM(MN, wc_q, Qb, nullptr, nullptr, SR_, 256, CM_, 0, 0, 1);
  GEMM(MN, wc_k, Kb, nullptr, nullptr, SR_, 256, CM_, 0, 0, 1);
  GEMM(MN, wc_v, Vb, nullptr, nullptr, SR_, 256, CM_, 0, 0, 1);
  k_attention<128><<<dim3(R_ * HM_, S_ / 64), 128, 0, stream>>>(
      Qb, Kb, Vb, nullptr, Ob, R_, HM_);
  GEMM(MN, wc_g, Gb, c_gb, Ob, SR_, 256, CM_, 2, 0, 1);
  GEMM(Gb, wc_o, T1, c_ob, nullptr, SR_, CM_, 256, 0, 0, 0);
  k_add_perm<<<PBLK, 256, 0, stream>>>(out_m, T1, S_, R_, CM_);

  // ================= MSA transition =================
  LN(out_m, t_lng, t_lnb, MN, SR_, CM_);
  GEMM(MN, wt_1, Hb, t_b1, nullptr, SR_, 4 * CM_, CM_, 1, 0, 1);
  GEMM(Hb, wt_2, out_m, t_b2, nullptr, SR_, CM_, 4 * CM_, 0, 1, 0);

  // ================= outer product mean =================
  LN(out_m, op_lng, op_lnb, MN, SR_, CM_);
  GEMM(MN, wo_a, Ab, op_ab, nullptr, SR_, 32, CM_, 0, 0, 1);
  GEMM(MN, wo_b, Bb, op_bb, nullptr, SR_, 32, CM_, 0, 0, 1);
  for (int i0 = 0; i0 < R_; i0 += 16) {
    k_opm1<<<dim3(16, R_ / 4), 128, 0, stream>>>(Ab, Bb, Pb, i0);
    GEMM(Pb, wo_o, out_z + (size_t)i0 * R_ * CZ_, op_ob, nullptr, 16 * R_,
         CZ_, 1024, 0, 1, 0);
  }

  // ================= triangle multiplications =================
  auto TRIMULT = [&](const TriP& p, const __bf16* wpa, const __bf16* wga,
                     const __bf16* wpb, const __bf16* wgb, const __bf16* wo,
                     const __bf16* wg, int outgoing) {
    LN(out_z, p.lng, p.lnb, ZN, RR_, CZ_);
    GEMM(ZN, wpa, T1, p.pab, nullptr, RR_, CZ_, CZ_, 0, 0, 0);
    GEMM(ZN, wga, Qb, p.gab, T1, RR_, CZ_, CZ_, 2, 0, 1);  // a (bf16)
    GEMM(ZN, wpb, T1, p.pbb, nullptr, RR_, CZ_, CZ_, 0, 0, 0);
    GEMM(ZN, wgb, Kb, p.gbb, T1, RR_, CZ_, CZ_, 2, 0, 1);  // b (bf16)
    k_tri<<<dim3(CZ_ / 4, R_ / 16, R_ / 16), 128, 0, stream>>>(Qb, Kb, T2,
                                                               outgoing);
    LN(T2, p.ln2g, p.ln2b, MN, RR_, CZ_);
    GEMM(ZN, wg, Ob, p.gb, nullptr, RR_, CZ_, CZ_, 2, 0, 0);  // gate (f32)
    GEMM(MN, wo, out_z, p.ob, Ob, RR_, CZ_, CZ_, 0, 1, 0);
  };
  TRIMULT(tro, wtro.pa, wtro.ga, wtro.pb, wtro.gb, wtro.o, wtro.g, 1);
  TRIMULT(tri, wtri.pa, wtri.ga, wtri.pb, wtri.gb, wtri.o, wtri.g, 0);

  // ================= triangle attentions =================
  auto TRIATT = [&](const float* zsrc, const TAP& p, const __bf16* wq,
                    const __bf16* wk, const __bf16* wv, const __bf16* wg,
                    const __bf16* wo, const __bf16* wb, float* dst,
                    int addto) {
    LN(zsrc, p.lng, p.lnb, ZN, RR_, CZ_);
    GEMMT(ZN, wb, BIAS, RR_, HZ_, CZ_);  // BIAS[h][q][k]
    GEMM(ZN, wq, Qb, nullptr, nullptr, RR_, 128, CZ_, 0, 0, 1);
    GEMM(ZN, wk, Kb, nullptr, nullptr, RR_, 128, CZ_, 0, 0, 1);
    GEMM(ZN, wv, Vb, nullptr, nullptr, RR_, 128, CZ_, 0, 0, 1);
    k_attention<256><<<dim3(R_ * HZ_, R_ / 64), 128, 0, stream>>>(
        Qb, Kb, Vb, BIAS, Ob, R_, HZ_);
    GEMM(ZN, wg, Gb, p.gb, Ob, RR_, 128, CZ_, 2, 0, 1);
    GEMM(Gb, wo, dst, p.ob, nullptr, RR_, CZ_, 128, 0, addto, 0);
  };
  TRIATT(out_z, tas, wtas.q, wtas.k, wtas.v, wtas.g, wtas.o, wtas.b, out_z,
         1);
  k_permute01<<<PBLK, 256, 0, stream>>>(MT, out_z, R_, R_, CZ_);
  TRIATT(MT, tae, wtae.q, wtae.k, wtae.v, wtae.g, wtae.o, wtae.b, T1, 0);
  k_add_perm<<<PBLK, 256, 0, stream>>>(out_z, T1, R_, R_, CZ_);

  // ================= pair transition =================
  LN(out_z, pt_lng, pt_lnb, ZN, RR_, CZ_);
  GEMM(ZN, wp_1, Hb, pt_b1, nullptr, RR_, 4 * CZ_, CZ_, 1, 0, 1);
  GEMM(Hb, wp_2, out_z, pt_b2, nullptr, RR_, CZ_, 4 * CZ_, 0, 1, 0);
}
